// DA_conv_64312840291006
// MI455X (gfx1250) — matrix operand order repaired and checked by reading, not compiled, not run on hardware
//
#include <hip/hip_runtime.h>
#include <math.h>

typedef __attribute__((ext_vector_type(16))) _Float16 v16h;
typedef __attribute__((ext_vector_type(8)))  float    v8f;

#define SLOPE 0.1f
#define B_   16
#define C_   64
#define H_   160
#define W_   160
#define TW   32          // tile width (pixels)
#define TH   2           // tile height (rows)
#define TPX  (TW*TH)     // 64 pixels per block
#define SF_CS 137        // sfeat channel stride in floats (4*34 + 1 pad)
#define TSTR  72         // t_lds pixel stride in halves (64 + 8 pad)

__device__ __forceinline__ float lrelu(float v) { return v > 0.f ? v : SLOPE * v; }

// Low 32 bits of a generic pointer to __shared__ = LDS byte offset
__device__ __forceinline__ unsigned lds_off(const void* p) {
    return (unsigned)(uintptr_t)p;
}

// ---------------- Stage 0: per-(b,c) 3x3 kernels + attention from deg ----------------
__global__ void __launch_bounds__(64) da_stage0(
    const float* __restrict__ deg, const float* __restrict__ W1,
    const float* __restrict__ W2,  const float* __restrict__ Wd1,
    const float* __restrict__ Wd2, float* __restrict__ kern_ws,
    float* __restrict__ att_ws)
{
    __shared__ float d_sh[64];
    __shared__ float h_sh[64];
    __shared__ float a_sh[8];
    const int b = blockIdx.x;
    const int t = threadIdx.x;

    d_sh[t] = deg[b * 64 + t];
    __syncthreads();

    float acc = 0.f;
    for (int i = 0; i < 64; ++i) acc += d_sh[i] * W1[t * 64 + i];
    h_sh[t] = lrelu(acc);

    if (t < 8) {
        float s = 0.f;
        for (int i = 0; i < 64; ++i) s += Wd1[t * 64 + i] * d_sh[i];
        a_sh[t] = lrelu(s);
    }
    __syncthreads();

    // k[b, oc, q] = sum_j h[j] * W2[(oc*9+q), j]
    for (int q = 0; q < 9; ++q) {
        float s = 0.f;
        const float* w2r = &W2[(t * 9 + q) * 64];
        for (int j = 0; j < 64; ++j) s += h_sh[j] * w2r[j];
        kern_ws[b * 576 + t * 9 + q] = s;
    }
    // att[b, o] = sigmoid(sum_r Wd2[o, r] * a[r])
    float s = 0.f;
    for (int r = 0; r < 8; ++r) s += Wd2[t * 8 + r] * a_sh[r];
    att_ws[b * 64 + t] = 1.f / (1.f + __expf(-s));
}

// ---------------- Stage 1: fused depthwise conv + WMMA channel mix + residual --------
__global__ void __launch_bounds__(256) da_main(
    const float* __restrict__ feat, const float* __restrict__ Wc,
    const float* __restrict__ bc,   const float* __restrict__ kern_ws,
    const float* __restrict__ att_ws, float* __restrict__ out)
{
    __shared__ float    sfeat[64 * SF_CS];     // 35072 B: feat tile + halo, fp32
    __shared__ _Float16 t_lds[TPX * TSTR];     //  9216 B: conv output, [px][ch] f16
    __shared__ _Float16 wa_lds[4 * 2 * 32 * 16]; // 8192 B: Wc pre-swizzled A fragments
    __shared__ float    kern_lds[64 * 9];      //  2304 B
    __shared__ float    att_lds[64];
    __shared__ float    bc_lds[64];

    const int x0   = blockIdx.x * TW;
    const int y0   = blockIdx.y * TH;
    const int b    = blockIdx.z;
    const int tid  = threadIdx.x;
    const int wave = tid >> 5;
    const int lane = tid & 31;

    const float* fb = feat + (size_t)b * 64 * H_ * W_;

    // Warm L2 for the next row-tile of this batch (global_prefetch_b8)
    if (y0 + TH < H_) {
        const float* pf = fb + ((size_t)(tid & 63) * H_ + (y0 + TH)) * W_ + x0;
        __builtin_prefetch(pf, 0, 3);
    }

    // Wc staged directly in WMMA A-fragment order:
    // wa_lds[((mt*2+s)*32 + lane)*16 + i]  = Wc[mt*16 + (lane&15)][s*32 + kl(lane,i)]
    // so each lane's fragment is one contiguous, 32B-aligned v16h.
    for (int idx = tid; idx < 4096; idx += 256) {
        const int i16 = idx & 15;
        const int ln  = (idx >> 4) & 31;
        const int s   = (idx >> 9) & 1;
        const int mt  = idx >> 10;
        const int lh  = ln >> 4;
        const int l15 = ln & 15;
        const int kl  = lh ? (i16 < 8 ? i16 + 8 : i16 + 16)
                           : (i16 < 8 ? i16 : i16 + 8);
        wa_lds[idx] = (_Float16)Wc[(mt * 16 + l15) * 64 + s * 32 + kl];
    }

    // Per-(b,c) 3x3 kernels: async copy global -> LDS (ASYNCcnt path)
    for (int i = tid; i < 576; i += 256) {
        const unsigned doff = lds_off(&kern_lds[i]);
        const float*   g    = &kern_ws[b * 576 + i];
        asm volatile("global_load_async_to_lds_b32 %0, %1, off"
                     :: "v"(doff), "v"(g) : "memory");
    }
    if (tid < 64) { att_lds[tid] = att_ws[b * 64 + tid]; bc_lds[tid] = bc[tid]; }

    // Feat tile + halo: 64 ch x 4 rows x 34 cols, zero-padded at image borders.
    // Interior rows (common case): async global->LDS copy for lanes 0..31,
    // checked scalar path for the 2-column tail and for boundary rows.
    for (int row = wave; row < 256; row += 8) {
        const int c  = row >> 2;
        const int r  = row & 3;
        const int gy = y0 - 1 + r;
        const bool rowok = (gy >= 0) && (gy < H_);
        const float* src = fb + ((size_t)c * H_ + gy) * W_;
        float* dst = &sfeat[c * SF_CS + r * 34];
        if (rowok && x0 > 0) {
            // gx = x0-1+lane in [x0-1, x0+30], always in range when x0 >= 32
            const unsigned doff = lds_off(dst + lane);
            const float*   g    = src + (x0 - 1 + lane);
            asm volatile("global_load_async_to_lds_b32 %0, %1, off"
                         :: "v"(doff), "v"(g) : "memory");
            if (lane < 2) {
                const int gx2 = x0 + 31 + lane;
                dst[32 + lane] = (gx2 < W_) ? src[gx2] : 0.f;
            }
        } else {
            const int gx = x0 - 1 + lane;
            float v = 0.f;
            if (rowok && gx >= 0 && gx < W_) v = src[gx];
            dst[lane] = v;
            if (lane < 2) {
                const int gx2 = x0 + 31 + lane;
                float v2 = 0.f;
                if (rowok && gx2 < W_) v2 = src[gx2];
                dst[32 + lane] = v2;
            }
        }
    }
    asm volatile("s_wait_asynccnt 0" ::: "memory");
    __syncthreads();

    // Depthwise 3x3 conv + leaky-ReLU -> f16 B-operand tile [px][ch]
    for (int it = 0; it < TPX * 64 / 256; ++it) {
        const int oi  = it * 256 + tid;
        const int ch  = oi & 63;
        const int px  = oi >> 6;          // 0..63
        const int row = px >> 5;
        const int col = px & 31;
        const float* sp = &sfeat[ch * SF_CS + row * 34 + col];
        const float* kp = &kern_lds[ch * 9];
        float acc = 0.f;
        #pragma unroll
        for (int dy = 0; dy < 3; ++dy)
            #pragma unroll
            for (int dx = 0; dx < 3; ++dx)
                acc += sp[dy * 34 + dx] * kp[dy * 3 + dx];
        t_lds[px * TSTR + ch] = (_Float16)lrelu(acc);
    }
    __syncthreads();

    // WMMA: D(64 x 64px) = Wc(64x64) x T(64x64px); 4 M-tiles x 4 N-tiles, 2 per wave
    const int mt    = wave & 3;
    const int m0    = mt * 16;
    const int npair = (wave >> 2) * 2;
    const int lhalf = lane >> 4;          // 0: lanes 0-15, 1: lanes 16-31
    const int l15   = lane & 15;

    // A fragments: single aligned 32B LDS read each (pre-swizzled above)
    v16h afrag[2];
    #pragma unroll
    for (int s = 0; s < 2; ++s) {
        afrag[s] = *(const v16h*)&wa_lds[(((mt * 2 + s) * 32) + lane) * 16];
    }

    #pragma unroll
    for (int nt = 0; nt < 2; ++nt) {
        const int n0 = (npair + nt) * 16;
        v8f acc = {};
        #pragma unroll
        for (int s = 0; s < 2; ++s) {
            // B fragment: halves k = 8h..8h+7 and 16+8h..16+8h+7 per lane (two ds_load_b128)
            v16h bfrag;
            const _Float16* bp = &t_lds[(n0 + l15) * TSTR + s * 32 + lhalf * 8];
            #pragma unroll
            for (int i = 0; i < 16; ++i) bfrag[i] = bp[i < 8 ? i : i + 8];
            acc = __builtin_amdgcn_wmma_f32_16x16x32_f16(
                false, afrag[s], false, bfrag, (short)0, acc, false, false);
        }
        // Epilogue: + bc + attention-scaled residual, coalesced fp32 stores
        const int px  = n0 + l15;
        const int row = px >> 5;
        const int col = px & 31;
        const int gy  = y0 + row;
        const int gx  = x0 + col;
        #pragma unroll
        for (int r = 0; r < 8; ++r) {
            const int o = m0 + r + lhalf * 8;
            const float res = acc[r] + bc_lds[o]
                            + sfeat[o * SF_CS + (row + 1) * 34 + (col + 1)] * att_lds[o];
            out[(((size_t)b * 64 + o) * H_ + gy) * W_ + gx] = res;
        }
    }
}

extern "C" void kernel_launch(void* const* d_in, const int* in_sizes, int n_in,
                              void* d_out, int out_size, void* d_ws, size_t ws_size,
                              hipStream_t stream) {
    const float* feat = (const float*)d_in[0];
    const float* deg  = (const float*)d_in[1];
    const float* W1   = (const float*)d_in[2];
    const float* W2   = (const float*)d_in[3];
    const float* Wc   = (const float*)d_in[4];
    const float* bc   = (const float*)d_in[5];
    const float* Wd1  = (const float*)d_in[6];
    const float* Wd2  = (const float*)d_in[7];
    float* out = (float*)d_out;

    float* kern_ws = (float*)d_ws;           // 16*64*9 floats
    float* att_ws  = kern_ws + 16 * 64 * 9;  // 16*64 floats

    da_stage0<<<16, 64, 0, stream>>>(deg, W1, W2, Wd1, Wd2, kern_ws, att_ws);

    dim3 grid(W_ / TW, H_ / TH, B_);
    da_main<<<grid, 256, 0, stream>>>(feat, Wc, bc, kern_ws, att_ws, out);
}
